// MLP_40321152974895
// MI455X (gfx1250) — matrix operand order repaired and checked by reading, not compiled, not run on hardware
//
#include <hip/hip_runtime.h>
#include <hip/hip_bf16.h>

typedef __attribute__((ext_vector_type(16))) _Float16 v16h;
typedef __attribute__((ext_vector_type(8)))  _Float16 v8h;
typedef __attribute__((ext_vector_type(4)))  _Float16 v4h;
typedef __attribute__((ext_vector_type(8)))  float    v8f;

#define GN    10000            // nodes
#define GE    320000           // edges
#define GD    128              // feature dim
#define GMAXN 64               // max in-degree (pad width)
#define GK1   (GMAXN * GD)     // 8192 = GEMM1 K
#define GH    1024             // hidden
#define GEPS  1e-5f

// async global->LDS copy of 16B; lds/global advance together via inst offset
#define ASYNC_B128(loff, goff, sbase, IMM)                                  \
  asm volatile("global_load_async_to_lds_b128 %0, %1, %2 offset:" IMM      \
               :: "v"(loff), "v"(goff), "s"(sbase) : "memory")

// ---------------- BN/bias folding:  y = x*alpha + beta ----------------
__global__ void prep_params_kernel(const float* __restrict__ b1, const float* __restrict__ g1,
                                   const float* __restrict__ be1, const float* __restrict__ rm1,
                                   const float* __restrict__ rv1,
                                   const float* __restrict__ b2, const float* __restrict__ g2,
                                   const float* __restrict__ be2, const float* __restrict__ rm2,
                                   const float* __restrict__ rv2,
                                   float* __restrict__ a1, float* __restrict__ bb1,
                                   float* __restrict__ a2, float* __restrict__ bb2) {
  int i = threadIdx.x;
  if (i < GH) {
    float s = g1[i] * rsqrtf(rv1[i] + GEPS);
    a1[i]  = s;
    bb1[i] = (b1[i] - rm1[i]) * s + be1[i];
  }
  if (i < GD) {
    float s = g2[i] * rsqrtf(rv2[i] + GEPS);
    a2[i]  = s;
    bb2[i] = (b2[i] - rm2[i]) * s + be2[i];
  }
}

// ------------- tiled transpose f32[rows,cols] -> f16[cols,rows] -------------
__global__ void transpose_f32_to_f16_kernel(const float* __restrict__ in,
                                            _Float16* __restrict__ out,
                                            int rows, int cols) {
  __shared__ float tile[32][33];
  int rb = blockIdx.x * 32;
  int cb = blockIdx.y * 32;
  int tx = threadIdx.x, ty = threadIdx.y;      // 32 x 8
#pragma unroll
  for (int i = 0; i < 4; ++i)
    tile[ty + i * 8][tx] = in[(size_t)(rb + ty + i * 8) * cols + (cb + tx)];
  __syncthreads();
#pragma unroll
  for (int i = 0; i < 4; ++i)
    out[(size_t)(cb + ty + i * 8) * rows + (rb + tx)] = (_Float16)tile[tx][ty + i * 8];
}

// ---------------- zero fill (float4 granularity) ----------------
__global__ void zero_kernel(float4* __restrict__ p, long n4) {
  long i = (long)blockIdx.x * blockDim.x + threadIdx.x;
  if (i < n4) p[i] = make_float4(0.f, 0.f, 0.f, 0.f);
}

// ------- scatter M[e,:] -> P[dest[e], pos(e), :] as f16 (pos via lower_bound) -------
__global__ void scatter_kernel(const float* __restrict__ M, const int* __restrict__ dest,
                               _Float16* __restrict__ P) {
  long idx = (long)blockIdx.x * blockDim.x + threadIdx.x;
  int e    = (int)(idx >> 5);
  int lane = (int)(idx & 31);
  if (e >= GE) return;
  int d = dest[e];
  int lo = 0, hi = e;                       // lower bound of d (dest sorted, dest[e]==d)
  while (lo < hi) {
    int mid = (lo + hi) >> 1;
    if (dest[mid] < d) lo = mid + 1; else hi = mid;
  }
  int pos = e - lo;
  if (pos >= GMAXN) return;
  const float* s = M + (size_t)e * GD + lane * 4;
  _Float16* dst  = P + ((size_t)d * GMAXN + pos) * GD + lane * 4;
  v4h o;
  o[0] = (_Float16)s[0]; o[1] = (_Float16)s[1];
  o[2] = (_Float16)s[2]; o[3] = (_Float16)s[3];
  *(v4h*)dst = o;
}

// ---------------- GEMM1: h = BN1(P @ W1 + b1), stored f16 ----------------
// Block: 256 thr = 8 waves, tile BM=64 x BN=256.
// B tile (256 cols x 32 K, f16) staged into LDS with async copies, double buffered.
// LDS row stride = 40 halves (80B) to spread banks; data occupies first 32 halves.
#define BROW 40
__global__ __launch_bounds__(256) void gemm1_kernel(const _Float16* __restrict__ P,
                                                    const _Float16* __restrict__ W1T,
                                                    const float* __restrict__ a1,
                                                    const float* __restrict__ bb1,
                                                    _Float16* __restrict__ Hout) {
  __shared__ __align__(128) _Float16 Bsh[2][256 * BROW];   // 2 x 20KB

  const int tid  = threadIdx.x;
  const int lane = tid & 31;
  const int w    = tid >> 5;
  const int wr   = w & 3;
  const int wc   = w >> 2;
  const int rowBase = blockIdx.x * 64 + wr * 16;
  const int colBase = blockIdx.y * 256;

  int arow = rowBase + (lane & 15);
  if (arow >= GN) arow = GN - 1;            // clamp; stores are guarded
  const _Float16* Ap = P + (size_t)arow * GK1 + ((lane >> 4) * 16);

  // async-copy bookkeeping: thread `tid` owns B row (colBase + tid), 64B per stage
  const unsigned ldsBase = (unsigned)(size_t)(&Bsh[0][0]) + (unsigned)tid * (BROW * 2);
  const unsigned gBase   = ((unsigned)(colBase + tid) * GK1) * 2;   // byte offset into W1T

  // fragment read offset inside LDS (halves)
  const int bfrag = (wc * 128 + (lane & 15)) * BROW + (lane >> 4) * 16;

  v8f acc[8] = {};

  // prologue: stage 0
  {
    unsigned goff = gBase;                  // k0 = 0
    unsigned loff = ldsBase;
    ASYNC_B128(loff, goff, W1T, "0");
    ASYNC_B128(loff, goff, W1T, "16");
    ASYNC_B128(loff, goff, W1T, "32");
    ASYNC_B128(loff, goff, W1T, "48");
  }

  for (int k0 = 0; k0 < GK1; k0 += 32) {
    const int buf = (k0 >> 5) & 1;
    const bool more = (k0 + 32) < GK1;
    if (more) {                             // prefetch stage k0+32 into other buffer
      unsigned goff = gBase + (unsigned)(k0 + 32) * 2;
      unsigned loff = ldsBase + (buf ^ 1) * (256 * BROW * 2);
      ASYNC_B128(loff, goff, W1T, "0");
      ASYNC_B128(loff, goff, W1T, "16");
      ASYNC_B128(loff, goff, W1T, "32");
      ASYNC_B128(loff, goff, W1T, "48");
      asm volatile("s_wait_asynccnt 4" ::: "memory");   // oldest 4 (this stage) done
    } else {
      asm volatile("s_wait_asynccnt 0" ::: "memory");
    }
    __syncthreads();                        // everyone's stage data visible in LDS

    v8h alo = *(const v8h*)(Ap + k0);       // K {0..7} / {16..23}
    v8h ahi = *(const v8h*)(Ap + k0 + 8);   // K {8..15} / {24..31}
    v16h a = __builtin_shufflevector(alo, ahi, 0,1,2,3,4,5,6,7,8,9,10,11,12,13,14,15);
    const _Float16* bp = &Bsh[buf][bfrag];
#pragma unroll
    for (int t = 0; t < 8; ++t) {
      v16h b = *(const v16h*)(bp + t * (16 * BROW));
      acc[t] = __builtin_amdgcn_wmma_f32_16x16x32_f16(false, a, false, b,
                                                      (short)0, acc[t], false, false);
    }
    __syncthreads();                        // all reads done before buffer is re-filled
  }

  const int rsub = (lane >> 4) * 8;
#pragma unroll
  for (int t = 0; t < 8; ++t) {
    int col = colBase + wc * 128 + t * 16 + (lane & 15);
    float al = a1[col], be = bb1[col];
#pragma unroll
    for (int r = 0; r < 8; ++r) {
      int row = rowBase + rsub + r;
      if (row < GN)
        Hout[(size_t)row * GH + col] = (_Float16)(acc[t][r] * al + be);
    }
  }
}

// ---------------- GEMM2: Mv = relu(BN2(h @ W2 + b2)), stored f32 ----------------
__global__ __launch_bounds__(256) void gemm2_kernel(const _Float16* __restrict__ Hin,
                                                    const _Float16* __restrict__ W2T,
                                                    const float* __restrict__ a2,
                                                    const float* __restrict__ bb2,
                                                    float* __restrict__ Mv) {
  const int lane = threadIdx.x & 31;
  const int w    = threadIdx.x >> 5;
  const int wr   = w & 3;
  const int wc   = w >> 2;
  const int rowBase = blockIdx.x * 64 + wr * 16;
  const int colBase = wc * 64;

  int arow = rowBase + (lane & 15);
  if (arow >= GN) arow = GN - 1;
  const _Float16* Ap = Hin + (size_t)arow * GH + ((lane >> 4) * 16);
  const int bcol0 = colBase + (lane & 15);
  const int boff  = (lane >> 4) * 16;

  v8f acc[4] = {};
  for (int k0 = 0; k0 < GH; k0 += 32) {
    v8h alo = *(const v8h*)(Ap + k0);
    v8h ahi = *(const v8h*)(Ap + k0 + 8);
    v16h a = __builtin_shufflevector(alo, ahi, 0,1,2,3,4,5,6,7,8,9,10,11,12,13,14,15);
#pragma unroll
    for (int t = 0; t < 4; ++t) {
      v16h b = *(const v16h*)(W2T + (size_t)(bcol0 + t * 16) * GH + k0 + boff);
      acc[t] = __builtin_amdgcn_wmma_f32_16x16x32_f16(false, a, false, b,
                                                      (short)0, acc[t], false, false);
    }
  }
  const int rsub = (lane >> 4) * 8;
#pragma unroll
  for (int t = 0; t < 4; ++t) {
    int col = colBase + t * 16 + (lane & 15);
    float al = a2[col], be = bb2[col];
#pragma unroll
    for (int r = 0; r < 8; ++r) {
      int row = rowBase + rsub + r;
      if (row < GN) {
        float v = acc[t][r] * al + be;
        Mv[(size_t)row * GD + col] = v > 0.f ? v : 0.f;
      }
    }
  }
}

// ---------------- out = Mv[src] - M[rev_index] ----------------
__global__ void final_kernel(const float4* __restrict__ Mv4, const float4* __restrict__ M4,
                             const int* __restrict__ srcIdx, const int* __restrict__ revIdx,
                             float4* __restrict__ out4) {
  long idx = (long)blockIdx.x * blockDim.x + threadIdx.x;
  if (idx >= (long)GE * 32) return;
  int e = (int)(idx >> 5);
  int j = (int)(idx & 31);
  int s  = srcIdx[e];
  int rv = revIdx[e];
  float4 a = Mv4[(size_t)s * 32 + j];
  float4 b = M4[(size_t)rv * 32 + j];
  out4[idx] = make_float4(a.x - b.x, a.y - b.y, a.z - b.z, a.w - b.w);
}

extern "C" void kernel_launch(void* const* d_in, const int* in_sizes, int n_in,
                              void* d_out, int out_size, void* d_ws, size_t ws_size,
                              hipStream_t stream) {
  (void)in_sizes; (void)n_in; (void)out_size; (void)ws_size;
  const float* M   = (const float*)d_in[0];
  const int*   ei  = (const int*)d_in[1];   // [2,E]: src then dest
  const int*   rev = (const int*)d_in[2];
  /* d_in[3] = dim_size (compile-time GN) */
  const float* W1  = (const float*)d_in[4];
  const float* b1  = (const float*)d_in[5];
  const float* g1  = (const float*)d_in[6];
  const float* be1 = (const float*)d_in[7];
  const float* rm1 = (const float*)d_in[8];
  const float* rv1 = (const float*)d_in[9];
  const float* W2  = (const float*)d_in[10];
  const float* b2  = (const float*)d_in[11];
  const float* g2  = (const float*)d_in[12];
  const float* be2 = (const float*)d_in[13];
  const float* rm2 = (const float*)d_in[14];
  const float* rv2 = (const float*)d_in[15];

  char* ws = (char*)d_ws;
  size_t off = 0;
  _Float16* P    = (_Float16*)(ws + off); off += (size_t)GN * GK1 * 2;  // 163.84 MB
  _Float16* W1T  = (_Float16*)(ws + off); off += (size_t)GH * GK1 * 2;  // 16.78 MB
  _Float16* W2T  = (_Float16*)(ws + off); off += (size_t)GD * GH * 2;   // 0.26 MB
  _Float16* Hbuf = (_Float16*)(ws + off); off += (size_t)GN * GH * 2;   // 20.48 MB
  float*    Mv   = (float*)(ws + off);    off += (size_t)GN * GD * 4;   // 5.12 MB
  float*    a1   = (float*)(ws + off);    off += GH * 4;
  float*    bb1  = (float*)(ws + off);    off += GH * 4;
  float*    a2   = (float*)(ws + off);    off += GD * 4;
  float*    bb2  = (float*)(ws + off);    off += GD * 4;

  const int* dest = ei + GE;

  prep_params_kernel<<<1, GH, 0, stream>>>(b1, g1, be1, rm1, rv1,
                                           b2, g2, be2, rm2, rv2, a1, bb1, a2, bb2);
  transpose_f32_to_f16_kernel<<<dim3(GK1 / 32, GH / 32), dim3(32, 8), 0, stream>>>(W1, W1T, GK1, GH);
  transpose_f32_to_f16_kernel<<<dim3(GH / 32, GD / 32), dim3(32, 8), 0, stream>>>(W2, W2T, GH, GD);

  long p4 = (long)GN * GK1 * 2 / 16;  // P in float4 units
  zero_kernel<<<(unsigned)((p4 + 255) / 256), 256, 0, stream>>>((float4*)P, p4);

  scatter_kernel<<<(GE * 32) / 256, 256, 0, stream>>>(M, dest, P);

  gemm1_kernel<<<dim3((GN + 63) / 64, GH / 256), 256, 0, stream>>>(P, W1T, a1, bb1, Hbuf);
  gemm2_kernel<<<dim3((GN + 63) / 64), 256, 0, stream>>>(Hbuf, W2T, a2, bb2, Mv);

  long nt = (long)GE * 32;
  final_kernel<<<(unsigned)((nt + 255) / 256), 256, 0, stream>>>((const float4*)Mv, (const float4*)M,
                                                                 ei, rev, (float4*)d_out);
}
